// NeuralFilterModule_34729105555539
// MI455X (gfx1250) — matrix operand order repaired and checked by reading, not compiled, not run on hardware
//
#include <hip/hip_runtime.h>

typedef _Float16 v8h  __attribute__((ext_vector_type(8)));
typedef _Float16 v16h __attribute__((ext_vector_type(16)));
typedef float    v8f  __attribute__((ext_vector_type(8)));

#define N_BLOCKS 10
#define TT 16384
#define BB 8
#define ROWS (BB*TT)      // 131072

// swizzled weight blob layout, f16 elements, per block:
#define TILE_EL 512                  // 32 lanes * 16 halves
#define SWZ_CONV0 0                  // 8 tiles: kt(2) x nt(4), tap0 (delayed)
#define SWZ_CONV1 (8*TILE_EL)        // tap1 (current)
#define SWZ_WF    (16*TILE_EL)       // 8 tiles: kt(4) x nt(2)
#define SWZ_WG    (24*TILE_EL)
#define SWZ_W1    (32*TILE_EL)       // 4 tiles: kt(1) x nt(4)
#define SWZ_W2P   (36*TILE_EL)       // 2 tiles: kt(2), N padded 2->16
#define TILES_PER_BLOCK 38
#define SWZ_PER_BLOCK (TILES_PER_BLOCK*TILE_EL)
#define SWZ_ID_OFF (N_BLOCKS*SWZ_PER_BLOCK)   // 4 identity tiles (64x64 I)

#define WMMA(a,b,c) __builtin_amdgcn_wmma_f32_16x16x32_f16(false,(a),false,(b),(short)0,(c),false,false)

// ---------------------------------------------------------------------------
// Swizzle all WMMA B-operands into per-lane-contiguous layout.
// B 32x16 (KxN) layout: lane L, element e -> K = (e&7) + 8*(L>=16) + 16*(e>>3), N = L%16.
// Also emits the padded w2p tiles and a 64x64 identity (for the residual).
// ---------------------------------------------------------------------------
__global__ void k_swizzle(const float* __restrict__ conv_w, const float* __restrict__ wf,
                          const float* __restrict__ wg, const float* __restrict__ w1,
                          const float* __restrict__ w2pbuf, _Float16* __restrict__ swz) {
  int tid = blockIdx.x * blockDim.x + threadIdx.x;
  const int nblk = N_BLOCKS * TILES_PER_BLOCK * 32;
  if (tid >= nblk + 4 * 32) return;
  if (tid >= nblk) {                  // identity tiles: nt 0..3, kt = nt>>1
    int idx = tid - nblk;
    int lane = idx & 31, it = idx >> 5;
    int laneLo = lane & 15, laneHi = lane >> 4;
    _Float16* dst = swz + SWZ_ID_OFF + it * TILE_EL + lane * 16;
    #pragma unroll
    for (int e = 0; e < 16; ++e) {
      int K = (it >> 1) * 32 + (e & 7) + laneHi * 8 + (e >> 3) * 16;
      int N = it * 16 + laneLo;
      dst[e] = (_Float16)((K == N) ? 1.f : 0.f);
    }
    return;
  }
  int lane = tid & 31;
  int tile = (tid >> 5) % TILES_PER_BLOCK;
  int blk  = tid / (TILES_PER_BLOCK * 32);
  int laneLo = lane & 15, laneHi = lane >> 4;
  _Float16* dst = swz + (size_t)blk * SWZ_PER_BLOCK + tile * TILE_EL + lane * 16;
  #pragma unroll
  for (int e = 0; e < 16; ++e) {
    int kk = (e & 7) + laneHi * 8 + (e >> 3) * 16;  // K within a 32-row K-tile
    float v;
    if (tile < 16) {                    // conv: tiles 0..7 tap0, 8..15 tap1
      int tap = tile >> 3;
      int t8 = tile & 7, kt = t8 >> 2, nt = t8 & 3;
      int K = kt * 32 + kk, N = nt * 16 + laneLo;
      v = conv_w[((size_t)(blk * 64 + K) * 64 + N) * 2 + tap];
    } else if (tile < 32) {             // wf (16..23) / wg (24..31)
      int isg = (tile >= 24);
      int t8 = (tile - 16) & 7, kt = t8 >> 1, nt = t8 & 1;
      int K = kt * 32 + kk, N = nt * 16 + laneLo;
      const float* W = isg ? wg : wf;
      v = W[(size_t)(blk * 128 + K) * 32 + N];
    } else if (tile < 36) {             // w1: K=32, nt 0..3
      int nt = tile - 32;
      int K = kk, N = nt * 16 + laneLo;
      v = w1[(size_t)(blk * 32 + K) * 64 + N];
    } else {                            // folded output weights, N padded to 16
      int kt = tile - 36;
      int K = kt * 32 + kk;
      v = (laneLo < 2) ? w2pbuf[blk * 128 + K * 2 + laneLo] : 0.f;
    }
    dst[e] = (_Float16)v;
  }
}

// ---------------------------------------------------------------------------
// Fold w2 @ p1w @ p2w -> w2p[blk][64][2]  and the constant bias 2-vector.
// ---------------------------------------------------------------------------
__global__ void k_fold(const float* __restrict__ w2, const float* __restrict__ b2,
                       const float* __restrict__ p1w, const float* __restrict__ p1b,
                       const float* __restrict__ p2w, const float* __restrict__ p2b,
                       float* __restrict__ w2p, float* __restrict__ pconst) {
  int tid = blockIdx.x * blockDim.x + threadIdx.x;
  if (tid < N_BLOCKS * 64 * 2) {
    int j = tid & 1, k = (tid >> 1) & 63, blk = tid >> 7;
    float s = 0.f;
    for (int m = 0; m < 128; ++m) {
      float q = 0.f;
      for (int n = 0; n < 16; ++n) q += p1w[m * 16 + n] * p2w[n * 2 + j];
      s += w2[(size_t)(blk * 64 + k) * 128 + m] * q;
    }
    w2p[tid] = s;
  } else if (tid < N_BLOCKS * 64 * 2 + 2) {
    int j = tid - N_BLOCKS * 64 * 2;
    float s = p2b[j];
    for (int n = 0; n < 16; ++n) s += p1b[n] * p2w[n * 2 + j];
    for (int m = 0; m < 128; ++m) {
      float bm = 0.f;
      for (int i = 0; i < N_BLOCKS; ++i) bm += b2[i * 128 + m];
      float q = 0.f;
      for (int n = 0; n < 16; ++n) q += p1w[m * 16 + n] * p2w[n * 2 + j];
      s += bm * q;
    }
    pconst[j] = s;
  }
}

// ---------------------------------------------------------------------------
// h0 = x @ cw + cb  (f16), zero acc2.
// ---------------------------------------------------------------------------
__global__ void k_init(const float* __restrict__ x, const float* __restrict__ cw,
                       const float* __restrict__ cb, _Float16* __restrict__ h0,
                       float* __restrict__ acc2) {
  int row = blockIdx.x * blockDim.x + threadIdx.x;
  if (row >= ROWS) return;
  float xv = x[row];
  #pragma unroll 8
  for (int n = 0; n < 64; ++n) h0[(size_t)row * 64 + n] = (_Float16)(xv * cw[n] + cb[n]);
  acc2[row * 2 + 0] = 0.f;
  acc2[row * 2 + 1] = 0.f;
}

// ---------------------------------------------------------------------------
// One WaveNet block, fully fused. 4 waves/WG, 16 rows per wave-tile,
// each WG loops over 8 tiles (grid 256 -> 512 rows/WG).
// Residual add and the folded 64->2 output projection are done with WMMA
// (identity B-tiles / padded w2p B-tiles) instead of scalar LDS traffic.
// ---------------------------------------------------------------------------
__global__ void __launch_bounds__(128)
k_block(const _Float16* __restrict__ hsrc, _Float16* __restrict__ hdst,
        const float* __restrict__ cf32, float* __restrict__ acc2,
        const _Float16* __restrict__ swzB, const _Float16* __restrict__ swzI,
        const float* __restrict__ conv_b, const float* __restrict__ bf,
        const float* __restrict__ bg, const float* __restrict__ b1, int dil)
{
  __shared__ __align__(32) _Float16 sW[SWZ_PER_BLOCK];  // 38 KB swizzled weights
  __shared__ __align__(32) _Float16 sI[4 * TILE_EL];    // 4 KB identity tiles
  __shared__ __align__(16) _Float16 sConv[4][16][64];   // conv output (z cols 64..127)
  __shared__ __align__(16) _Float16 sH[4][16][64];      // hres tile (for store + w2p A)
  __shared__ __align__(16) _Float16 sG[4][16][32];      // gated activation

  const int tid = threadIdx.x;
  const int wave = tid >> 5;
  const int lane = tid & 31;
  const int laneLo = lane & 15, laneHi = lane >> 4;

  // cooperative load of swizzled weights + identity tiles
  {
    const v8h* src = (const v8h*)swzB;
    v8h* dst = (v8h*)sW;
    for (int idx = tid; idx < SWZ_PER_BLOCK / 8; idx += 128) dst[idx] = src[idx];
    const v8h* srcI = (const v8h*)swzI;
    v8h* dstI = (v8h*)sI;
    for (int idx = tid; idx < (4 * TILE_EL) / 8; idx += 128) dstI[idx] = srcI[idx];
  }
  __syncthreads();

  const float cbv[4] = {conv_b[laneLo], conv_b[16 + laneLo], conv_b[32 + laneLo], conv_b[48 + laneLo]};
  const float bfv[2] = {bf[laneLo], bf[16 + laneLo]};
  const float bgv[2] = {bg[laneLo], bg[16 + laneLo]};
  const float b1v[4] = {b1[laneLo], b1[16 + laneLo], b1[32 + laneLo], b1[48 + laneLo]};

  const int rowsPerWG = ROWS / gridDim.x;
  const int wgBase = blockIdx.x * rowsPerWG;

  for (int tileBase = wgBase + wave * 16; tileBase < wgBase + rowsPerWG; tileBase += 64) {
    const int arow = tileBase + laneLo;        // A-operand row for this lane
    const int t = arow & (TT - 1);             // time index within batch

    // ---------------- dilated conv: acc = h[t-d]@W0 + h[t]@W1 + b ----------
    v8f acc[4];
    #pragma unroll
    for (int nt = 0; nt < 4; ++nt)
      #pragma unroll
      for (int r = 0; r < 8; ++r) acc[nt][r] = cbv[nt];

    {   // tap 0 (delayed, zero-padded per batch)
      const bool valid = (t >= dil);
      const int srow = valid ? (arow - dil) : arow;
      const v16h az = {};
      #pragma unroll
      for (int kt = 0; kt < 2; ++kt) {
        const _Float16* p = hsrc + (size_t)srow * 64 + kt * 32 + laneHi * 8;
        v16h a;
        *(v8h*)&a       = *(const v8h*)p;
        *((v8h*)&a + 1) = *(const v8h*)(p + 16);
        if (!valid) a = az;
        #pragma unroll
        for (int nt = 0; nt < 4; ++nt) {
          const v16h b = *(const v16h*)&sW[SWZ_CONV0 + (kt * 4 + nt) * TILE_EL + lane * 16];
          acc[nt] = WMMA(a, b, acc[nt]);
        }
      }
    }
    v16h aT1[2];   // tap-1 A operands, kept live for the identity residual
    #pragma unroll
    for (int kt = 0; kt < 2; ++kt) {
      const _Float16* p = hsrc + (size_t)arow * 64 + kt * 32 + laneHi * 8;
      *(v8h*)&aT1[kt]       = *(const v8h*)p;
      *((v8h*)&aT1[kt] + 1) = *(const v8h*)(p + 16);
      #pragma unroll
      for (int nt = 0; nt < 4; ++nt) {
        const v16h b = *(const v16h*)&sW[SWZ_CONV1 + (kt * 4 + nt) * TILE_EL + lane * 16];
        acc[nt] = WMMA(aT1[kt], b, acc[nt]);
      }
    }
    // scatter conv result (C-layout -> row-major f16 LDS)
    #pragma unroll
    for (int nt = 0; nt < 4; ++nt)
      #pragma unroll
      for (int r = 0; r < 8; ++r)
        sConv[wave][laneHi * 8 + r][nt * 16 + laneLo] = (_Float16)acc[nt][r];
    asm volatile("s_wait_dscnt 0x0" ::: "memory");

    // ---------------- filter & gate: z=[c|conv], 128 -> 32 each -----------
    v8f fA[2], gA[2];
    #pragma unroll
    for (int nt = 0; nt < 2; ++nt)
      #pragma unroll
      for (int r = 0; r < 8; ++r) { fA[nt][r] = bfv[nt]; gA[nt][r] = bgv[nt]; }

    #pragma unroll
    for (int kt = 0; kt < 4; ++kt) {
      v16h a;
      if (kt < 2) {   // context half of z, converted f32 -> f16 on the fly
        const float* pc = cf32 + (size_t)arow * 64 + kt * 32 + laneHi * 8;
        const v8f c0 = *(const v8f*)pc;
        const v8f c1 = *(const v8f*)(pc + 16);
        #pragma unroll
        for (int e = 0; e < 8; ++e) { a[e] = (_Float16)c0[e]; a[8 + e] = (_Float16)c1[e]; }
      } else {        // conv half of z from LDS
        const _Float16* pz = &sConv[wave][laneLo][(kt - 2) * 32 + laneHi * 8];
        *(v8h*)&a       = *(const v8h*)pz;
        *((v8h*)&a + 1) = *(const v8h*)(pz + 16);
      }
      #pragma unroll
      for (int nt = 0; nt < 2; ++nt) {
        const v16h bF = *(const v16h*)&sW[SWZ_WF + (kt * 2 + nt) * TILE_EL + lane * 16];
        const v16h bG = *(const v16h*)&sW[SWZ_WG + (kt * 2 + nt) * TILE_EL + lane * 16];
        fA[nt] = WMMA(a, bF, fA[nt]);
        gA[nt] = WMMA(a, bG, gA[nt]);
      }
    }
    // gated activation, scatter to LDS
    #pragma unroll
    for (int nt = 0; nt < 2; ++nt)
      #pragma unroll
      for (int r = 0; r < 8; ++r) {
        const float fv = tanhf(fA[nt][r]);
        const float gv = 1.f / (1.f + expf(-gA[nt][r]));
        sG[wave][laneHi * 8 + r][nt * 16 + laneLo] = (_Float16)(fv * gv);
      }
    asm volatile("s_wait_dscnt 0x0" ::: "memory");

    // ------- linear1 (32 -> 64) + residual via identity WMMA --------------
    v8f hacc[4];
    #pragma unroll
    for (int nt = 0; nt < 4; ++nt)
      #pragma unroll
      for (int r = 0; r < 8; ++r) hacc[nt][r] = b1v[nt];
    {
      const _Float16* pg = &sG[wave][laneLo][laneHi * 8];
      v16h ga;
      *(v8h*)&ga       = *(const v8h*)pg;
      *((v8h*)&ga + 1) = *(const v8h*)(pg + 16);
      #pragma unroll
      for (int nt = 0; nt < 4; ++nt) {
        const v16h b = *(const v16h*)&sW[SWZ_W1 + nt * TILE_EL + lane * 16];
        hacc[nt] = WMMA(ga, b, hacc[nt]);
      }
    }
    #pragma unroll
    for (int nt = 0; nt < 4; ++nt) {     // hacc[nt] += h @ I   (residual)
      const v16h bI = *(const v16h*)&sI[nt * TILE_EL + lane * 16];
      hacc[nt] = WMMA(aT1[nt >> 1], bI, hacc[nt]);
    }
    // scatter hres (f32 accum -> f16 row-major LDS)
    #pragma unroll
    for (int nt = 0; nt < 4; ++nt)
      #pragma unroll
      for (int r = 0; r < 8; ++r)
        sH[wave][laneHi * 8 + r][nt * 16 + laneLo] = (_Float16)hacc[nt][r];
    asm volatile("s_wait_dscnt 0x0" ::: "memory");

    // hres -> global (vectorized)
    {
      const _Float16* srcp = &sH[wave][laneLo][laneHi * 32];
      _Float16* dstp = hdst + (size_t)(tileBase + laneLo) * 64 + laneHi * 32;
      #pragma unroll
      for (int q = 0; q < 4; ++q) *(v8h*)(dstp + q * 8) = *(const v8h*)(srcp + q * 8);
    }
    // folded output projection hres @ w2p (N padded to 16) via WMMA
    {
      v8f dacc = {};
      #pragma unroll
      for (int kt = 0; kt < 2; ++kt) {
        const _Float16* ph = &sH[wave][laneLo][kt * 32 + laneHi * 8];
        v16h a;
        *(v8h*)&a       = *(const v8h*)ph;
        *((v8h*)&a + 1) = *(const v8h*)(ph + 16);
        const v16h b = *(const v16h*)&sW[SWZ_W2P + kt * TILE_EL + lane * 16];
        dacc = WMMA(a, b, dacc);
      }
      if (laneLo < 2) {      // only N=0,1 are real output columns
        #pragma unroll
        for (int r = 0; r < 8; ++r)
          acc2[(tileBase + laneHi * 8 + r) * 2 + laneLo] += dacc[r];
      }
    }
  }
}

// ---------------------------------------------------------------------------
// out = x * exp(p1) + p0
// ---------------------------------------------------------------------------
__global__ void k_final(const float* __restrict__ x, const float* __restrict__ acc2,
                        const float* __restrict__ pconst, float* __restrict__ out) {
  int row = blockIdx.x * blockDim.x + threadIdx.x;
  if (row >= ROWS) return;
  const float p0 = acc2[row * 2 + 0] + pconst[0];
  const float p1 = acc2[row * 2 + 1] + pconst[1];
  out[row] = x[row] * expf(p1) + p0;
}

// ---------------------------------------------------------------------------
extern "C" void kernel_launch(void* const* d_in, const int* in_sizes, int n_in,
                              void* d_out, int out_size, void* d_ws, size_t ws_size,
                              hipStream_t stream) {
  const float* x      = (const float*)d_in[0];
  const float* c      = (const float*)d_in[1];
  const float* cw     = (const float*)d_in[2];
  const float* cb     = (const float*)d_in[3];
  const float* conv_w = (const float*)d_in[4];
  const float* conv_b = (const float*)d_in[5];
  const float* wf     = (const float*)d_in[6];
  const float* bf     = (const float*)d_in[7];
  const float* wg     = (const float*)d_in[8];
  const float* bg     = (const float*)d_in[9];
  const float* w1     = (const float*)d_in[10];
  const float* b1     = (const float*)d_in[11];
  const float* w2     = (const float*)d_in[12];
  const float* b2     = (const float*)d_in[13];
  const float* p1w    = (const float*)d_in[14];
  const float* p1b    = (const float*)d_in[15];
  const float* p2w    = (const float*)d_in[16];
  const float* p2b    = (const float*)d_in[17];

  char* ws = (char*)d_ws;
  _Float16* hA   = (_Float16*)(ws);                                   // 16 MB
  _Float16* hB   = (_Float16*)(ws + (size_t)ROWS * 64 * 2);           // 16 MB
  float*    acc2 = (float*)(ws + (size_t)2 * ROWS * 64 * 2);          // 1 MB
  char* p = ws + (size_t)2 * ROWS * 64 * 2 + (size_t)ROWS * 2 * 4;
  _Float16* swz    = (_Float16*)p;  p += (size_t)(N_BLOCKS * SWZ_PER_BLOCK + 4 * TILE_EL) * 2;
  float*    w2p    = (float*)p;     p += (size_t)N_BLOCKS * 64 * 2 * 4;
  float*    pconst = (float*)p;

  k_fold<<<(N_BLOCKS * 64 * 2 + 2 + 255) / 256, 256, 0, stream>>>(w2, b2, p1w, p1b, p2w, p2b, w2p, pconst);
  k_swizzle<<<(N_BLOCKS * TILES_PER_BLOCK * 32 + 4 * 32 + 255) / 256, 256, 0, stream>>>(conv_w, wf, wg, w1, w2p, swz);
  k_init<<<ROWS / 256, 256, 0, stream>>>(x, cw, cb, hA, acc2);

  for (int i = 0; i < N_BLOCKS; ++i) {
    const _Float16* hs = (i & 1) ? hB : hA;
    _Float16*       hd = (i & 1) ? hA : hB;
    k_block<<<256, 128, 0, stream>>>(hs, hd, c, acc2,
        swz + (size_t)i * SWZ_PER_BLOCK, swz + SWZ_ID_OFF,
        conv_b + i * 64, bf + i * 32, bg + i * 32, b1 + i * 64, 1 << i);
  }

  k_final<<<ROWS / 256, 256, 0, stream>>>(x, acc2, pconst, (float*)d_out);
}
